// ExpertFFNGroupedMM_78099685310878
// MI455X (gfx1250) — compile-verified, matrix operand order checked by reading, not run on hardware
//
#include <hip/hip_runtime.h>
#include <stdint.h>

// MoE expert FFN (grouped GEMM + GLU) for gfx1250 / MI455X.
// route -> zgather(f16, padded) -> weight f32->f16 convert (one pass)
//       -> gemm1 (async-to-LDS staging, f16 WMMA, fused silu-GLU, f16 out)
//       -> gemm2 (async-to-LDS staging, f16 WMMA, scatter f32 out)

#define NEXP   8
#define TOPK   2
#define NTOKS  2048
#define DM     1024
#define DFFN   1024
#define ALIGNP 16
#define NROWS  (NTOKS * TOPK)            // 4096 routed rows
#define PMAX   (NROWS + NEXP * ALIGNP)   // 4224 padded slots
#define MAX_TILES (PMAX / 64 + NEXP)     // 74 per-expert 64-row M tiles

typedef __attribute__((ext_vector_type(16))) _Float16 v16h;
typedef __attribute__((ext_vector_type(8)))  _Float16 v8h;
typedef __attribute__((ext_vector_type(4)))  _Float16 v4h;
typedef __attribute__((ext_vector_type(8)))  float    v8f;

// ---- CDNA5 async copy: global -> LDS, 16B per lane, tracked by ASYNCcnt ----
__device__ inline void cp_async_b128(uint32_t lds_off, const void* sbase,
                                     uint32_t voff_bytes) {
  asm volatile("global_load_async_to_lds_b128 %0, %1, %2"
               :: "v"(lds_off), "v"(voff_bytes), "s"(sbase) : "memory");
}
__device__ inline uint32_t lds_addr(const void* p) {
  return (uint32_t)(uint64_t)p;   // low 32 bits of shared addr = LDS byte addr
}

// ---------------------------------------------------------------------------
// Routing: slot_flat[slot] = flat routed-row id (-1 = padding). Segmented
// per-expert tile table (64-row tiles, 16-aligned group starts).
// ---------------------------------------------------------------------------
__global__ __launch_bounds__(256) void moe_route(
    const int* __restrict__ indices, const int* __restrict__ counts,
    int* __restrict__ slot_flat, int* __restrict__ tile_base,
    int* __restrict__ tile_rows, int* __restrict__ tile_e) {
  __shared__ int pstart[NEXP + 1];
  __shared__ int cnt[NEXP];
  const int tid = threadIdx.x;
  if (tid == 0) {
    int acc = 0;
    for (int e = 0; e < NEXP; ++e) {
      pstart[e] = acc;
      acc += ((counts[e] + ALIGNP - 1) / ALIGNP) * ALIGNP;
    }
    pstart[NEXP] = acc;
  }
  if (tid < NEXP) cnt[tid] = 0;
  for (int s = tid; s < PMAX; s += 256) slot_flat[s] = -1;
  __syncthreads();
  for (int j = tid; j < NROWS; j += 256) {
    int e = indices[j];
    int r = atomicAdd(&cnt[e], 1);
    slot_flat[pstart[e] + r] = j;
  }
  if (tid == 0) {
    int t = 0;
    for (int e = 0; e < NEXP; ++e) {
      for (int b = pstart[e]; b < pstart[e + 1]; b += 64) {
        tile_base[t] = b;
        int rem = pstart[e + 1] - b;
        tile_rows[t] = rem < 64 ? rem : 64;
        tile_e[t] = e;
        ++t;
      }
    }
    for (; t < MAX_TILES; ++t) { tile_base[t] = 0; tile_rows[t] = 0; tile_e[t] = -1; }
  }
}

// ---------------------------------------------------------------------------
// Gather x into padded-slot order as f16 (zeros for pad slots). One block/row.
// ---------------------------------------------------------------------------
__global__ __launch_bounds__(256) void moe_zgather(
    const float* __restrict__ x, const int* __restrict__ slot_flat,
    _Float16* __restrict__ zbuf) {
  const int s = blockIdx.x;
  const int tid = threadIdx.x;
  const int j = slot_flat[s];
  v4h h = {};
  if (j >= 0) {
    float4 v = *(const float4*)(x + (size_t)(j / TOPK) * DM + tid * 4);
    h[0] = (_Float16)v.x; h[1] = (_Float16)v.y;
    h[2] = (_Float16)v.z; h[3] = (_Float16)v.w;
  }
  *(v4h*)(zbuf + (size_t)s * DM + tid * 4) = h;
}

// One-pass weight conversion f32 -> f16 (row-major layout preserved).
__global__ __launch_bounds__(256) void f32_to_f16(
    const float* __restrict__ src, _Float16* __restrict__ dst, int n4) {
  int i = blockIdx.x * 256 + threadIdx.x;
  if (i < n4) {
    float4 v = *(const float4*)(src + (size_t)i * 4);
    v4h h = {(_Float16)v.x, (_Float16)v.y, (_Float16)v.z, (_Float16)v.w};
    *(v4h*)(dst + (size_t)i * 4) = h;
  }
}

// Build 16x32 f16 WMMA fragment from one LDS row (32 halves).
__device__ inline v16h load_frag(const _Float16* __restrict__ row, int kb) {
  v8h lo = *(const v8h*)(row + kb);
  v8h hi = *(const v8h*)(row + kb + 16);
  v16h f;
#pragma unroll
  for (int i = 0; i < 8; ++i) { f[i] = lo[i]; f[i + 8] = hi[i]; }
  return f;
}

// ---------------------------------------------------------------------------
// GEMM1: hact = y * silu(gate); block 64 rows x 128 cols (y and gate).
// Double-buffered async-to-LDS staging: 5 b128 copies/thread/step.
// ---------------------------------------------------------------------------
__global__ __launch_bounds__(256) void moe_gemm1(
    const _Float16* __restrict__ zbuf, const _Float16* __restrict__ fc1h,
    const int* __restrict__ tile_base, const int* __restrict__ tile_rows,
    const int* __restrict__ tile_e, _Float16* __restrict__ hact) {
  const int t = blockIdx.x;
  const int e = tile_e[t];
  if (e < 0) return;
  const int base = tile_base[t];
  const int rows = tile_rows[t];
  const int n0   = blockIdx.y * 128;

  __shared__ _Float16 As[2][64][32];
  __shared__ _Float16 Bs[2][256][32];   // 0..127 y rows, 128..255 gate rows

  const int tid  = threadIdx.x;
  const int wave = tid >> 5, lane = tid & 31;
  const int mw = wave & 1, nw = wave >> 1;
  const int fr = lane & 15;
  const int kb = (lane >> 4) * 8;

  const _Float16* Za = zbuf + (size_t)base * DM;
  const _Float16* Wa = fc1h + (size_t)e * (2 * DFFN) * DM;

  // A: one 16B chunk per thread per k-step.
  const int ar  = tid >> 2;
  const int ac8 = (tid & 3) * 8;
  const uint32_t a_goff = (uint32_t)(ar * DM + ac8) * 2;
  const uint32_t a_lds[2] = { lds_addr(&As[0][ar][ac8]), lds_addr(&As[1][ar][ac8]) };

  // B: four 16B chunks per thread per k-step.
  uint32_t b_goff[4], b_lds[2][4];
#pragma unroll
  for (int j = 0; j < 4; ++j) {
    int lin = tid + j * 256;
    int r = lin >> 2;                       // 0..255
    int c8 = (lin & 3) * 8;
    int wr = (r < 128) ? (n0 + r) : (DFFN + n0 + (r - 128));
    b_goff[j] = (uint32_t)(wr * DM + c8) * 2;
    b_lds[0][j] = lds_addr(&Bs[0][r][c8]);
    b_lds[1][j] = lds_addr(&Bs[1][r][c8]);
  }

  v8f zacc = {};
  v8f cy[2][2], cg[2][2];
#pragma unroll
  for (int mi = 0; mi < 2; ++mi)
#pragma unroll
    for (int ni = 0; ni < 2; ++ni) { cy[mi][ni] = zacc; cg[mi][ni] = zacc; }

  const int S = DM / 32;
  // Prologue: step 0 -> buffer 0.
  cp_async_b128(a_lds[0], Za, a_goff);
#pragma unroll
  for (int j = 0; j < 4; ++j) cp_async_b128(b_lds[0][j], Wa, b_goff[j]);

  for (int i = 0; i < S; ++i) {
    const int b = i & 1;
    if (i + 1 < S) {
      const uint32_t kadd = (uint32_t)((i + 1) * 64);  // 32 halves = 64 B
      cp_async_b128(a_lds[b ^ 1], Za, a_goff + kadd);
#pragma unroll
      for (int j = 0; j < 4; ++j)
        cp_async_b128(b_lds[b ^ 1][j], Wa, b_goff[j] + kadd);
      asm volatile("s_wait_asynccnt 0x5" ::: "memory");  // step i retired
    } else {
      asm volatile("s_wait_asynccnt 0x0" ::: "memory");
    }
    __syncthreads();   // all waves' copies into buffer b landed

    v16h a[2], by[2], bg[2];
#pragma unroll
    for (int mi = 0; mi < 2; ++mi)
      a[mi] = load_frag(&As[b][32 * mw + 16 * mi + fr][0], kb);
#pragma unroll
    for (int ni = 0; ni < 2; ++ni) {
      by[ni] = load_frag(&Bs[b][32 * nw + 16 * ni + fr][0], kb);
      bg[ni] = load_frag(&Bs[b][128 + 32 * nw + 16 * ni + fr][0], kb);
    }
#pragma unroll
    for (int mi = 0; mi < 2; ++mi)
#pragma unroll
      for (int ni = 0; ni < 2; ++ni) {
        cy[mi][ni] = __builtin_amdgcn_wmma_f32_16x16x32_f16(
            false, a[mi], false, by[ni], (short)0, cy[mi][ni], false, false);
        cg[mi][ni] = __builtin_amdgcn_wmma_f32_16x16x32_f16(
            false, a[mi], false, bg[ni], (short)0, cg[mi][ni], false, false);
      }
    __syncthreads();   // buffer b free for step i+2's copies
  }

  // Fused GLU epilogue. C layout: vgpr v -> M = v + 8*(lane>=16), N = lane&15.
  const int rb = (lane >> 4) * 8;
#pragma unroll
  for (int mi = 0; mi < 2; ++mi)
#pragma unroll
    for (int ni = 0; ni < 2; ++ni) {
      int col = n0 + 32 * nw + 16 * ni + fr;
#pragma unroll
      for (int i = 0; i < 8; ++i) {
        int rt = 32 * mw + 16 * mi + rb + i;
        if (rt < rows) {   // mask rows owned by the next expert's tile
          float g = cg[mi][ni][i];
          float s = g / (1.0f + __expf(-g));
          hact[(size_t)(base + rt) * DFFN + col] = (_Float16)(cy[mi][ni][i] * s);
        }
      }
    }
}

// ---------------------------------------------------------------------------
// GEMM2: zo = hact @ fc2^T, scattered to flat row order. 3 copies/thread/step.
// ---------------------------------------------------------------------------
__global__ __launch_bounds__(256) void moe_gemm2(
    const _Float16* __restrict__ hact, const _Float16* __restrict__ fc2h,
    const int* __restrict__ slot_flat, const int* __restrict__ tile_base,
    const int* __restrict__ tile_rows, const int* __restrict__ tile_e,
    float* __restrict__ out) {
  const int t = blockIdx.x;
  const int e = tile_e[t];
  if (e < 0) return;
  const int base = tile_base[t];
  const int rows = tile_rows[t];
  const int n0   = blockIdx.y * 128;

  __shared__ _Float16 As[2][64][32];
  __shared__ _Float16 Bs[2][128][32];
  __shared__ int jr[64];

  const int tid  = threadIdx.x;
  const int wave = tid >> 5, lane = tid & 31;
  const int mw = wave & 1, nw = wave >> 1;
  const int fr = lane & 15;
  const int kb = (lane >> 4) * 8;

  if (tid < 64) jr[tid] = (tid < rows) ? slot_flat[base + tid] : -1;

  const _Float16* Ha = hact + (size_t)base * DFFN;
  const _Float16* Wa = fc2h + (size_t)e * DM * DFFN;

  const int ar  = tid >> 2;
  const int ac8 = (tid & 3) * 8;
  const uint32_t a_goff = (uint32_t)(ar * DFFN + ac8) * 2;
  const uint32_t a_lds[2] = { lds_addr(&As[0][ar][ac8]), lds_addr(&As[1][ar][ac8]) };

  uint32_t b_goff[2], b_lds[2][2];
#pragma unroll
  for (int j = 0; j < 2; ++j) {
    int lin = tid + j * 256;
    int r = lin >> 2;                       // 0..127
    int c8 = (lin & 3) * 8;
    b_goff[j] = (uint32_t)((n0 + r) * DFFN + c8) * 2;
    b_lds[0][j] = lds_addr(&Bs[0][r][c8]);
    b_lds[1][j] = lds_addr(&Bs[1][r][c8]);
  }

  v8f zacc = {};
  v8f c[2][2];
#pragma unroll
  for (int mi = 0; mi < 2; ++mi)
#pragma unroll
    for (int ni = 0; ni < 2; ++ni) c[mi][ni] = zacc;

  const int S = DFFN / 32;
  cp_async_b128(a_lds[0], Ha, a_goff);
#pragma unroll
  for (int j = 0; j < 2; ++j) cp_async_b128(b_lds[0][j], Wa, b_goff[j]);

  for (int i = 0; i < S; ++i) {
    const int b = i & 1;
    if (i + 1 < S) {
      const uint32_t kadd = (uint32_t)((i + 1) * 64);
      cp_async_b128(a_lds[b ^ 1], Ha, a_goff + kadd);
#pragma unroll
      for (int j = 0; j < 2; ++j)
        cp_async_b128(b_lds[b ^ 1][j], Wa, b_goff[j] + kadd);
      asm volatile("s_wait_asynccnt 0x3" ::: "memory");
    } else {
      asm volatile("s_wait_asynccnt 0x0" ::: "memory");
    }
    __syncthreads();

    v16h a[2], bfrag[2];
#pragma unroll
    for (int mi = 0; mi < 2; ++mi)
      a[mi] = load_frag(&As[b][32 * mw + 16 * mi + fr][0], kb);
#pragma unroll
    for (int ni = 0; ni < 2; ++ni)
      bfrag[ni] = load_frag(&Bs[b][32 * nw + 16 * ni + fr][0], kb);
#pragma unroll
    for (int mi = 0; mi < 2; ++mi)
#pragma unroll
      for (int ni = 0; ni < 2; ++ni)
        c[mi][ni] = __builtin_amdgcn_wmma_f32_16x16x32_f16(
            false, a[mi], false, bfrag[ni], (short)0, c[mi][ni], false, false);
    __syncthreads();
  }

  const int rb = (lane >> 4) * 8;
#pragma unroll
  for (int mi = 0; mi < 2; ++mi)
#pragma unroll
    for (int ni = 0; ni < 2; ++ni) {
      int col = n0 + 32 * nw + 16 * ni + fr;
#pragma unroll
      for (int i = 0; i < 8; ++i) {
        int rt = 32 * mw + 16 * mi + rb + i;
        int j = (rt < rows) ? jr[rt] : -1;
        if (j >= 0) out[(size_t)j * DM + col] = c[mi][ni][i];
      }
    }
}

extern "C" void kernel_launch(void* const* d_in, const int* in_sizes, int n_in,
                              void* d_out, int out_size, void* d_ws, size_t ws_size,
                              hipStream_t stream) {
  const float* x       = (const float*)d_in[0];
  const float* fc1     = (const float*)d_in[1];
  const float* fc2     = (const float*)d_in[2];
  const int*   indices = (const int*)d_in[3];
  const int*   counts  = (const int*)d_in[4];
  float* out = (float*)d_out;

  char* ws = (char*)d_ws;
  int* slot_flat = (int*)ws;
  int* tile_base = slot_flat + PMAX;
  int* tile_rows = tile_base + MAX_TILES;
  int* tile_e    = tile_rows + MAX_TILES;

  size_t off = 32768;
  _Float16* zbuf = (_Float16*)(ws + off); off += (size_t)PMAX * DM * 2;
  _Float16* hact = (_Float16*)(ws + off); off += (size_t)PMAX * DFFN * 2;
  _Float16* fc1h = (_Float16*)(ws + off); off += (size_t)NEXP * 2 * DFFN * DM * 2;
  _Float16* fc2h = (_Float16*)(ws + off);

  const int fc1n4 = NEXP * 2 * DFFN * DM / 4;   // 4,194,304
  const int fc2n4 = NEXP * DM * DFFN / 4;       // 2,097,152

  moe_route<<<1, 256, 0, stream>>>(indices, counts, slot_flat,
                                   tile_base, tile_rows, tile_e);
  moe_zgather<<<PMAX, 256, 0, stream>>>(x, slot_flat, zbuf);
  f32_to_f16<<<fc1n4 / 256, 256, 0, stream>>>(fc1, fc1h, fc1n4);
  f32_to_f16<<<fc2n4 / 256, 256, 0, stream>>>(fc2, fc2h, fc2n4);
  moe_gemm1<<<dim3(MAX_TILES, DFFN / 128), 256, 0, stream>>>(
      zbuf, fc1h, tile_base, tile_rows, tile_e, hact);
  moe_gemm2<<<dim3(MAX_TILES, DM / 128), 256, 0, stream>>>(
      hact, fc2h, slot_flat, tile_base, tile_rows, tile_e, out);
}
